// SVIT_63204738728631
// MI455X (gfx1250) — 16-bit and 8-bit matrix operand order repaired and checked by reading, compiled, not run on hardware
//
#include <hip/hip_runtime.h>
#include <cstdint>


typedef __attribute__((ext_vector_type(16))) _Float16 v16h;
typedef __attribute__((ext_vector_type(8)))  _Float16 v8h;
typedef __attribute__((ext_vector_type(8)))  float    v8f;
typedef __attribute__((ext_vector_type(8)))  int      v8i;
typedef __attribute__((ext_vector_type(4)))  int      v4i;

#define BN_EPS 1e-5f

#if defined(__has_builtin)
# if __has_builtin(__builtin_amdgcn_global_load_async_to_lds_b128) && \
     __has_builtin(__builtin_amdgcn_s_wait_asynccnt)
#  define HAVE_ASYNC_LDS 1
# endif
#endif
#ifndef HAVE_ASYNC_LDS
# define HAVE_ASYNC_LDS 0
#endif

__device__ __forceinline__ v16h cat8(v8h lo, v8h hi) {
  return __builtin_shufflevector(lo, hi, 0,1,2,3,4,5,6,7,8,9,10,11,12,13,14,15);
}

#if HAVE_ASYNC_LDS
__device__ __forceinline__ void async_cp16(const _Float16* g, _Float16* l) {
  __builtin_amdgcn_global_load_async_to_lds_b128(
      (__attribute__((address_space(1))) v4i*)(void*)g,
      (__attribute__((address_space(3))) v4i*)(void*)l, 0, 0);
}
#endif

__global__ void fold_linear_k(const float* __restrict__ w, const float* __restrict__ lb,
                              const float* __restrict__ g, const float* __restrict__ bb,
                              const float* __restrict__ m, const float* __restrict__ vv,
                              _Float16* __restrict__ wf, float* __restrict__ bias,
                              int O, int K)
{
  long i = (long)blockIdx.x * blockDim.x + threadIdx.x;
  if (i >= (long)O * K) return;
  int o = (int)(i / K);
  float inv = g[o] * rsqrtf(vv[o] + BN_EPS);
  wf[i] = (_Float16)(w[i] * inv);
  if ((i % K) == 0) bias[o] = lb[o] * inv + bb[o] - m[o] * inv;
}

__global__ void fold_conv_k(const float* __restrict__ w,
                            const float* __restrict__ g, const float* __restrict__ bb,
                            const float* __restrict__ m, const float* __restrict__ vv,
                            _Float16* __restrict__ wf, float* __restrict__ bias,
                            int O, int I, int Ip)
{
  long i = (long)blockIdx.x * blockDim.x + threadIdx.x;
  if (i >= 9L * O * Ip) return;
  int ci  = (int)(i % Ip);
  int o   = (int)((i / Ip) % O);
  int tap = (int)(i / ((long)Ip * O));
  float inv = g[o] * rsqrtf(vv[o] + BN_EPS);
  wf[i] = (ci < I) ? (_Float16)(w[((long)o * I + ci) * 9 + tap] * inv) : (_Float16)0.f;
  if (ci == 0 && tap == 0) bias[o] = bb[o] - m[o] * inv;
}

__global__ void prep_input_k(const float* __restrict__ X, _Float16* __restrict__ A,
                             int B, int T, int C, int H, int W, int Cp)
{
  long i = (long)blockIdx.x * blockDim.x + threadIdx.x;
  long total = (long)T * B * H * W;
  if (i >= total) return;
  int w  = (int)(i % W);
  int h  = (int)((i / W) % H);
  int tb = (int)(i / ((long)H * W));
  int b = tb % B, t = tb / B;
  _Float16* dst = A + i * Cp;
  for (int c = 0; c < Cp; ++c)
    dst[c] = (c < C) ? (_Float16)X[((((long)b * T + t) * C + c) * H + h) * W + w]
                     : (_Float16)0.f;
}

__global__ __launch_bounds__(256) void gemm_f16_wmma_k(
    const _Float16* __restrict__ A, const _Float16* __restrict__ Wm,
    const float* __restrict__ bias, _Float16* __restrict__ C,
    int M, int N, int K)
{
#if HAVE_ASYNC_LDS
  __shared__ _Float16 As[2][16][32];
#endif
  int lane = threadIdx.x & 31, wave = threadIdx.x >> 5;
  int half = lane >> 4, r = lane & 15;
  int mt = blockIdx.x;
  int nt = blockIdx.y * 8 + wave;
  const _Float16* arow = A  + (size_t)(mt * 16 + r) * K;
  const _Float16* wrow = Wm + (size_t)(nt * 64 + r) * K;
  const size_t wstep = (size_t)16 * K;
  v8f acc0 = {}, acc1 = {}, acc2 = {}, acc3 = {};
  const int nk = K >> 5;
#if HAVE_ASYNC_LDS
  const _Float16* ag = arow + half * 8;
  if (wave == 0) {
    async_cp16(ag,      &As[0][r][half * 8]);
    async_cp16(ag + 16, &As[0][r][16 + half * 8]);
  }
#endif
  for (int ks = 0; ks < nk; ++ks) {
    int k = ks << 5;
#if HAVE_ASYNC_LDS
    if (wave == 0) {
      if (ks + 1 < nk) {
        const _Float16* an = ag + (size_t)(ks + 1) * 32;
        int nb = (ks + 1) & 1;
        async_cp16(an,      &As[nb][r][half * 8]);
        async_cp16(an + 16, &As[nb][r][16 + half * 8]);
        __builtin_amdgcn_s_wait_asynccnt(2);
      } else {
        __builtin_amdgcn_s_wait_asynccnt(0);
      }
    }
    __syncthreads();
    int cb = ks & 1;
    v8h alo = *(const v8h*)&As[cb][r][half * 8];
    v8h ahi = *(const v8h*)&As[cb][r][16 + half * 8];
#else
    v8h alo = *(const v8h*)(arow + k + half * 8);
    v8h ahi = *(const v8h*)(arow + k + 16 + half * 8);
#endif
    v16h af = cat8(alo, ahi);
    const _Float16* wk = wrow + k + half * 8;
    v16h b0 = cat8(*(const v8h*)(wk), *(const v8h*)(wk + 16));
    v16h b1 = cat8(*(const v8h*)(wk + wstep), *(const v8h*)(wk + wstep + 16));
    v16h b2 = cat8(*(const v8h*)(wk + 2 * wstep), *(const v8h*)(wk + 2 * wstep + 16));
    v16h b3 = cat8(*(const v8h*)(wk + 3 * wstep), *(const v8h*)(wk + 3 * wstep + 16));
    if (k + 128 < K) __builtin_prefetch(wrow + k + 128, 0, 1);
    acc0 = __builtin_amdgcn_wmma_f32_16x16x32_f16(false, af, false, b0, (short)0, acc0, false, false);
    acc1 = __builtin_amdgcn_wmma_f32_16x16x32_f16(false, af, false, b1, (short)0, acc1, false, false);
    acc2 = __builtin_amdgcn_wmma_f32_16x16x32_f16(false, af, false, b2, (short)0, acc2, false, false);
    acc3 = __builtin_amdgcn_wmma_f32_16x16x32_f16(false, af, false, b3, (short)0, acc3, false, false);
#if HAVE_ASYNC_LDS
    __syncthreads();
#endif
  }
  int m0 = mt * 16 + half * 8;
  v8f* accs[4] = {&acc0, &acc1, &acc2, &acc3};
  for (int j = 0; j < 4; ++j) {
    int n = nt * 64 + j * 16 + r;
    float bn = bias ? bias[n] : 0.f;
    for (int q = 0; q < 8; ++q)
      C[(size_t)(m0 + q) * N + n] = (_Float16)((*accs[j])[q] + bn);
  }
}

__global__ __launch_bounds__(256) void conv3x3_wmma_k(
    const _Float16* __restrict__ X, const _Float16* __restrict__ Wt,
    const float* __restrict__ bias, _Float16* __restrict__ Y,
    int TB, int H, int W, int Cin, int Cout)
{
  int lane = threadIdx.x & 31, wave = threadIdx.x >> 5;
  int ntiles = Cout >> 6;
  long mtiles = ((long)TB * H * W) >> 4;
  long tile = (long)blockIdx.x * (blockDim.x >> 5) + wave;
  if (tile >= mtiles * ntiles) return;
  int nt  = (int)(tile % ntiles);
  long pt = tile / ntiles;
  int tpr = W >> 4;
  int w0  = (int)(pt % tpr) * 16;
  long rowi = pt / tpr;
  int hh = (int)(rowi % H);
  int tb = (int)(rowi / H);
  int half = lane >> 4, r = lane & 15;
  int wl = w0 + r;
  const size_t wstep = (size_t)16 * Cin;
  v8f acc0 = {}, acc1 = {}, acc2 = {}, acc3 = {};
  for (int tap = 0; tap < 9; ++tap) {
    int dh = tap / 3 - 1, dw = tap % 3 - 1;
    int h2 = hh + dh, w2 = wl + dw;
    bool valid = (h2 >= 0) && (h2 < H) && (w2 >= 0) && (w2 < W);
    const _Float16* px = X + ((long)(tb * H + h2) * W + w2) * Cin + half * 8;
    const _Float16* pw = Wt + ((size_t)tap * Cout + nt * 64 + r) * Cin + half * 8;
    for (int k = 0; k < Cin; k += 32) {
      v8h z = {};
      v8h alo = valid ? *(const v8h*)(px + k)      : z;
      v8h ahi = valid ? *(const v8h*)(px + k + 16) : z;
      v16h af = cat8(alo, ahi);
      v16h b0 = cat8(*(const v8h*)(pw + k), *(const v8h*)(pw + k + 16));
      v16h b1 = cat8(*(const v8h*)(pw + k + wstep), *(const v8h*)(pw + k + wstep + 16));
      v16h b2 = cat8(*(const v8h*)(pw + k + 2 * wstep), *(const v8h*)(pw + k + 2 * wstep + 16));
      v16h b3 = cat8(*(const v8h*)(pw + k + 3 * wstep), *(const v8h*)(pw + k + 3 * wstep + 16));
      acc0 = __builtin_amdgcn_wmma_f32_16x16x32_f16(false, af, false, b0, (short)0, acc0, false, false);
      acc1 = __builtin_amdgcn_wmma_f32_16x16x32_f16(false, af, false, b1, (short)0, acc1, false, false);
      acc2 = __builtin_amdgcn_wmma_f32_16x16x32_f16(false, af, false, b2, (short)0, acc2, false, false);
      acc3 = __builtin_amdgcn_wmma_f32_16x16x32_f16(false, af, false, b3, (short)0, acc3, false, false);
    }
  }
  v8f* accs[4] = {&acc0, &acc1, &acc2, &acc3};
  for (int j = 0; j < 4; ++j) {
    int n = nt * 64 + j * 16 + r;
    float bn = bias[n];
    for (int q = 0; q < 8; ++q) {
      int wq = w0 + q + half * 8;
      Y[((size_t)(tb * H + hh) * W + wq) * Cout + n] = (_Float16)((*accs[j])[q] + bn);
    }
  }
}

__global__ void lif_k(const _Float16* __restrict__ X, _Float16* __restrict__ Of,
                      unsigned char* __restrict__ Ou8, int T, long S, float vth,
                      int trans, int B, int Ntok, int Dtot)
{
  long i = (long)blockIdx.x * blockDim.x + threadIdx.x;
  if (i >= S) return;
  long c = 0, n = 0, b = 0;
  if (trans) { c = i % Dtot; n = (i / Dtot) % Ntok; b = i / ((long)Dtot * Ntok); }
  float v = 0.f;
  for (int t = 0; t < T; ++t) {
    float x = (float)X[(long)t * S + i];
    v = 0.5f * (v + x);
    bool sp = (v >= vth);
    if (Of)  Of[(long)t * S + i] = sp ? (_Float16)1.f : (_Float16)0.f;
    if (Ou8) {
      long oi = trans ? ((((long)t * B + b) * Dtot + c) * Ntok + n)
                      : ((long)t * S + i);
      Ou8[oi] = sp ? (unsigned char)1 : (unsigned char)0;
    }
    if (sp) v = 0.f;
  }
}

__global__ void maxpool_k(const _Float16* __restrict__ X, _Float16* __restrict__ Y,
                          int TB, int H, int W, int C)
{
  int Ho = H >> 1, Wo = W >> 1;
  long total = (long)TB * Ho * Wo * C;
  long i = (long)blockIdx.x * blockDim.x + threadIdx.x;
  if (i >= total) return;
  int c  = (int)(i % C);
  int wo = (int)((i / C) % Wo);
  int ho = (int)((i / ((long)C * Wo)) % Ho);
  int tb = (int)(i / ((long)C * Wo * Ho));
  float mx = -1e30f;
  for (int dy = 0; dy < 3; ++dy)
    for (int dx = 0; dx < 3; ++dx) {
      int h = ho * 2 - 1 + dy, w = wo * 2 - 1 + dx;
      if (h >= 0 && h < H && w >= 0 && w < W)
        mx = fmaxf(mx, (float)X[((long)(tb * H + h) * W + w) * C + c]);
    }
  Y[i] = (_Float16)mx;
}

__global__ void add2_k(const _Float16* __restrict__ A, const _Float16* __restrict__ Bv,
                       _Float16* __restrict__ O, long n)
{
  long i = (long)blockIdx.x * blockDim.x + threadIdx.x;
  if (i >= n) return;
  O[i] = (_Float16)((float)A[i] + (float)Bv[i]);
}

__global__ __launch_bounds__(128) void ssa_attn_wmma_k(
    const unsigned char* __restrict__ Q, const unsigned char* __restrict__ Kc,
    const unsigned char* __restrict__ Vt, _Float16* __restrict__ Y,
    int N, int Hd, float scale)
{
  __shared__ unsigned char lds[16 * 64];
  int lane = threadIdx.x & 31, wave = threadIdx.x >> 5;
  int qt = blockIdx.x;
  int tb = blockIdx.y / Hd, h = blockIdx.y % Hd;
  int half = lane >> 4, r = lane & 15;
  int Dt = Hd * 64;

  v8i aq;
  {
    const unsigned char* p = Q + (size_t)(tb * N + qt * 16 + r) * Dt + h * 64 + half * 16;
    for (int j = 0; j < 4; ++j) {
      uint2 u = *(const uint2*)(p + (j & 1) * 8 + (j >> 1) * 32);
      aq[2 * j] = (int)u.x; aq[2 * j + 1] = (int)u.y;
    }
  }

  v8i acc = {};
  for (int mc = 0; mc < N / 64; ++mc) {
    int key0 = mc * 64 + wave * 16;
    v8i bk;
    {
      const unsigned char* p = Kc + (size_t)(tb * N + key0 + r) * Dt + h * 64;
      uint4 b0 = *(const uint4*)(p + half * 16);
      uint4 b1 = *(const uint4*)(p + 32 + half * 16);
      bk[0] = (int)b0.x; bk[1] = (int)b0.y; bk[2] = (int)b0.z; bk[3] = (int)b0.w;
      bk[4] = (int)b1.x; bk[5] = (int)b1.y; bk[6] = (int)b1.z; bk[7] = (int)b1.w;
    }
    v8i zero = {};
    v8i s1 = __builtin_amdgcn_wmma_i32_16x16x64_iu8(false, aq, false, bk,
                                                    zero, false, false);
    for (int q = 0; q < 8; ++q)
      lds[(q + half * 8) * 64 + wave * 16 + r] = (unsigned char)s1[q];
    __syncthreads();

    v8i aa;
    {
      const unsigned char* p = lds + r * 64 + half * 16;
      for (int j = 0; j < 4; ++j) {
        uint2 u = *(const uint2*)(p + (j & 1) * 8 + (j >> 1) * 32);
        aa[2 * j] = (int)u.x; aa[2 * j + 1] = (int)u.y;
      }
    }
    v8i bv;
    {
      const unsigned char* p =
          Vt + ((size_t)(tb * Hd + h) * 64 + wave * 16 + r) * N + mc * 64;
      uint4 b0 = *(const uint4*)(p + half * 16);
      uint4 b1 = *(const uint4*)(p + 32 + half * 16);
      bv[0] = (int)b0.x; bv[1] = (int)b0.y; bv[2] = (int)b0.z; bv[3] = (int)b0.w;
      bv[4] = (int)b1.x; bv[5] = (int)b1.y; bv[6] = (int)b1.z; bv[7] = (int)b1.w;
    }
    acc = __builtin_amdgcn_wmma_i32_16x16x64_iu8(false, aa, false, bv,
                                                 acc, false, false);
    __syncthreads();
  }
  int col = h * 64 + wave * 16 + r;
  for (int q = 0; q < 8; ++q) {
    int qq = qt * 16 + q + half * 8;
    Y[(size_t)(tb * N + qq) * Dt + col] = (_Float16)(scale * (float)acc[q]);
  }
}

__global__ void token_mean_k(const _Float16* __restrict__ X, float* __restrict__ Mn,
                             int T, int B, int N, int D)
{
  int i = blockIdx.x * blockDim.x + threadIdx.x;
  if (i >= B * D) return;
  int b = i / D, d = i % D;
  float s = 0.f;
  for (int t = 0; t < T; ++t)
    for (int n = 0; n < N; ++n)
      s += (float)X[((size_t)((t * B + b) * N + n)) * D + d];
  Mn[i] = s / (float)(T * N);
}

__global__ void classifier_k(const float* __restrict__ Mn, const float* __restrict__ Wc,
                             const float* __restrict__ bc, float* __restrict__ out,
                             int B, int D, int NC)
{
  int i = blockIdx.x * blockDim.x + threadIdx.x;
  if (i >= B * NC) return;
  int b = i / NC, c = i % NC;
  float s = bc[c];
  for (int d = 0; d < D; ++d) s += Mn[b * D + d] * Wc[c * D + d];
  out[i] = s;
}

extern "C" void kernel_launch(void* const* d_in, const int* in_sizes, int n_in,
                              void* d_out, int out_size, void* d_ws, size_t ws_size,
                              hipStream_t stream)
{
  (void)in_sizes; (void)n_in; (void)out_size; (void)ws_size;
  constexpr int B = 2, T = 4, TB = 8, H = 128, W = 128, D = 512, N = 1024,
                Hd = 8, NC = 10;

  auto F = [&](int i) { return (const float*)d_in[i]; };
  const float* Xin = F(0);
  const int iC0 = 1, iBN0 = 2, iC1 = 6, iBN1 = 7, iC2 = 11, iBN2 = 12,
            iC3 = 16, iBN3 = 17, iRPE = 21, iRPEBN = 22;
  auto blk = [&](int bi) { return 26 + bi * 36; };
  const int iCLSw = 26 + 4 * 36, iCLSb = iCLSw + 1;

  char* wsb = (char*)d_ws;
  size_t off = 0;
  auto alloc = [&](size_t bytes) -> char* {
    char* p = wsb + off;
    off += (bytes + 255) & ~(size_t)255;
    return p;
  };
  _Float16* c0w = (_Float16*)alloc(9L *  64 *  32 * 2);  float* c0b = (float*)alloc( 64 * 4);
  _Float16* c1w = (_Float16*)alloc(9L * 128 *  64 * 2);  float* c1b = (float*)alloc(128 * 4);
  _Float16* c2w = (_Float16*)alloc(9L * 256 * 128 * 2);  float* c2b = (float*)alloc(256 * 4);
  _Float16* c3w = (_Float16*)alloc(9L * 512 * 256 * 2);  float* c3b = (float*)alloc(512 * 4);
  _Float16* rpw = (_Float16*)alloc(9L * 512 * 512 * 2);  float* rpb = (float*)alloc(512 * 4);
  _Float16 *qw[4], *kw[4], *vw[4], *pw[4], *f1w[4], *f2w[4];
  float    *qb[4], *kb[4], *vb[4], *pb[4], *f1b[4], *f2b[4];
  for (int bi = 0; bi < 4; ++bi) {
    qw[bi]  = (_Float16*)alloc((size_t)D * D * 2);      qb[bi]  = (float*)alloc(D * 4);
    kw[bi]  = (_Float16*)alloc((size_t)D * D * 2);      kb[bi]  = (float*)alloc(D * 4);
    vw[bi]  = (_Float16*)alloc((size_t)D * D * 2);      vb[bi]  = (float*)alloc(D * 4);
    pw[bi]  = (_Float16*)alloc((size_t)D * D * 2);      pb[bi]  = (float*)alloc(D * 4);
    f1w[bi] = (_Float16*)alloc((size_t)4 * D * D * 2);  f1b[bi] = (float*)alloc(4 * D * 4);
    f2w[bi] = (_Float16*)alloc((size_t)D * 4 * D * 2);  f2b[bi] = (float*)alloc(D * 4);
  }
  _Float16* act0 = (_Float16*)alloc((size_t)TB * H * W * 32 * 2);
  _Float16* big  = (_Float16*)alloc((size_t)TB * H * W * 256 * 2);
  _Float16* sp   = (_Float16*)alloc((size_t)TB * H * W * 256 * 2);
  _Float16* pool = (_Float16*)alloc((size_t)TB * 64 * 64 * 256 * 2);
  _Float16* xbuf = (_Float16*)alloc((size_t)TB * N * D * 2);
  unsigned char* qu8 = (unsigned char*)alloc((size_t)TB * N * D);
  unsigned char* ku8 = (unsigned char*)alloc((size_t)TB * N * D);
  unsigned char* vu8 = (unsigned char*)alloc((size_t)TB * N * D);
  float* meanbuf = (float*)alloc((size_t)B * D * 4);

  auto foldConv = [&](int iw, int ibn, _Float16* wf, float* bias, int O, int I, int Ip) {
    long n = 9L * O * Ip;
    fold_conv_k<<<(unsigned)((n + 255) / 256), 256, 0, stream>>>(
        F(iw), F(ibn + 0), F(ibn + 1), F(ibn + 2), F(ibn + 3), wf, bias, O, I, Ip);
  };
  auto foldLin = [&](int ibase, _Float16* wf, float* bias, int O, int K) {
    long n = (long)O * K;
    fold_linear_k<<<(unsigned)((n + 255) / 256), 256, 0, stream>>>(
        F(ibase + 0), F(ibase + 1), F(ibase + 2), F(ibase + 3), F(ibase + 4),
        F(ibase + 5), wf, bias, O, K);
  };
  auto gemm = [&](const _Float16* A, const _Float16* Wm, const float* bias,
                  _Float16* C, int M, int Nn, int K) {
    dim3 g(M / 16, Nn / 512);
    gemm_f16_wmma_k<<<g, 256, 0, stream>>>(A, Wm, bias, C, M, Nn, K);
  };
  auto conv = [&](const _Float16* X, const _Float16* Wt, const float* bias,
                  _Float16* Y, int h, int w, int cin, int cout) {
    long tiles = ((long)TB * h * w / 16) * (cout / 64);
    conv3x3_wmma_k<<<(unsigned)((tiles + 7) / 8), 256, 0, stream>>>(X, Wt, bias, Y, TB, h, w, cin, cout);
  };
  auto lif = [&](const _Float16* X, _Float16* Of, unsigned char* Ou8, long S,
                 float vth, int trans) {
    lif_k<<<(unsigned)((S + 255) / 256), 256, 0, stream>>>(X, Of, Ou8, T, S, vth, trans, B, N, D);
  };

  foldConv(iC0, iBN0, c0w, c0b,  64,   2,  32);
  foldConv(iC1, iBN1, c1w, c1b, 128,  64,  64);
  foldConv(iC2, iBN2, c2w, c2b, 256, 128, 128);
  foldConv(iC3, iBN3, c3w, c3b, 512, 256, 256);
  foldConv(iRPE, iRPEBN, rpw, rpb, 512, 512, 512);
  for (int bi = 0; bi < 4; ++bi) {
    int b0 = blk(bi);
    foldLin(b0 + 0,  qw[bi],  qb[bi],  D, D);
    foldLin(b0 + 6,  kw[bi],  kb[bi],  D, D);
    foldLin(b0 + 12, vw[bi],  vb[bi],  D, D);
    foldLin(b0 + 18, pw[bi],  pb[bi],  D, D);
    foldLin(b0 + 24, f1w[bi], f1b[bi], 4 * D, D);
    foldLin(b0 + 30, f2w[bi], f2b[bi], D, 4 * D);
  }

  {
    long tot = (long)TB * H * W;
    prep_input_k<<<(unsigned)((tot + 255) / 256), 256, 0, stream>>>(Xin, act0, B, T, 2, H, W, 32);
  }
  conv(act0, c0w, c0b, big, H, W, 32, 64);
  lif(big, sp, nullptr, (long)B * H * W * 64, 1.f, 0);
  conv(sp, c1w, c1b, big, H, W, 64, 128);
  lif(big, sp, nullptr, (long)B * H * W * 128, 1.f, 0);
  conv(sp, c2w, c2b, big, H, W, 128, 256);
  lif(big, sp, nullptr, (long)B * H * W * 256, 1.f, 0);
  {
    long tot = (long)TB * 64 * 64 * 256;
    maxpool_k<<<(unsigned)((tot + 255) / 256), 256, 0, stream>>>(sp, pool, TB, H, W, 256);
  }
  conv(pool, c3w, c3b, big, 64, 64, 256, 512);
  lif(big, sp, nullptr, (long)B * 64 * 64 * 512, 1.f, 0);
  {
    long tot = (long)TB * 32 * 32 * 512;
    maxpool_k<<<(unsigned)((tot + 255) / 256), 256, 0, stream>>>(sp, pool, TB, 64, 64, 512);
  }
  conv(pool, rpw, rpb, big, 32, 32, 512, 512);
  lif(big, sp, nullptr, (long)B * 32 * 32 * 512, 1.f, 0);
  {
    long tot = (long)TB * N * D;
    add2_k<<<(unsigned)((tot + 255) / 256), 256, 0, stream>>>(sp, pool, xbuf, tot);
  }

  const long Stok = (long)B * N * D;
  const long Atok = (long)TB * N * D;
  for (int bi = 0; bi < 4; ++bi) {
    gemm(xbuf, qw[bi], qb[bi], big, TB * N, D, D);
    lif(big, nullptr, qu8, Stok, 1.f, 0);
    gemm(xbuf, kw[bi], kb[bi], big, TB * N, D, D);
    lif(big, nullptr, ku8, Stok, 1.f, 0);
    gemm(xbuf, vw[bi], vb[bi], big, TB * N, D, D);
    lif(big, nullptr, vu8, Stok, 1.f, 1);
    {
      dim3 g(N / 16, TB * Hd);
      ssa_attn_wmma_k<<<g, 128, 0, stream>>>(qu8, ku8, vu8, big, N, Hd, 0.125f);
    }
    lif(big, sp, nullptr, Stok, 0.5f, 0);
    gemm(sp, pw[bi], pb[bi], big, TB * N, D, D);
    lif(big, sp, nullptr, Stok, 1.f, 0);
    add2_k<<<(unsigned)((Atok + 255) / 256), 256, 0, stream>>>(xbuf, sp, xbuf, Atok);
    gemm(xbuf, f1w[bi], f1b[bi], big, TB * N, 4 * D, D);
    lif(big, sp, nullptr, (long)B * N * 4 * D, 1.f, 0);
    gemm(sp, f2w[bi], f2b[bi], big, TB * N, D, 4 * D);
    lif(big, sp, nullptr, Stok, 1.f, 0);
    add2_k<<<(unsigned)((Atok + 255) / 256), 256, 0, stream>>>(xbuf, sp, xbuf, Atok);
  }

  token_mean_k<<<(B * D + 255) / 256, 256, 0, stream>>>(xbuf, meanbuf, T, B, N, D);
  classifier_k<<<1, 64, 0, stream>>>(meanbuf, F(iCLSw), F(iCLSb), (float*)d_out, B, D, NC);
}
